// HybridMultiheadAttention_18107582120107
// MI455X (gfx1250) — compile-verified, matrix operand order checked by reading, not run on hardware
//
#include <hip/hip_runtime.h>
#include <hip/hip_bf16.h>

typedef __attribute__((ext_vector_type(16))) __bf16 v16bf;
typedef __attribute__((ext_vector_type(8)))  float  v8f;

constexpr int SEQ = 2048;
constexpr int EMB = 1024;
constexpr int NH  = 16;
constexpr int HDIM = 64;
constexpr int CAUSAL_HEADS = 4;
constexpr int WINDOW = 63;
constexpr int PSTRIDE = 56;   // bf16 elems per LDS row: 112B, 16B-aligned, conflict-free
constexpr float FLOORF = -3.402823466e38f;   // finite running-max floor

union Frag {
    v16bf v;
    float4 f4[2];
    __bf16 e[16];
};

__device__ __forceinline__ unsigned short f2bf(float f) {
    __bf16 h = (__bf16)f;                       // native v_cvt path
    return __builtin_bit_cast(unsigned short, h);
}

// A/B 16-bit fragment layout (ISA 7.12.2): lane l owns row/col (l&15); its 16
// elements cover K = base + (l>>4)*8 + [0..7] and K = base+16 + (l>>4)*8 + [0..7].
__device__ __forceinline__ v16bf load_frag_f32(const float* row, int k0) {
    Frag fr;
    float4 a = ((const float4*)(row + k0))[0];
    float4 b = ((const float4*)(row + k0))[1];
    float4 c = ((const float4*)(row + k0 + 16))[0];
    float4 d = ((const float4*)(row + k0 + 16))[1];
    fr.e[0] = (__bf16)a.x;  fr.e[1] = (__bf16)a.y;  fr.e[2] = (__bf16)a.z;  fr.e[3] = (__bf16)a.w;
    fr.e[4] = (__bf16)b.x;  fr.e[5] = (__bf16)b.y;  fr.e[6] = (__bf16)b.z;  fr.e[7] = (__bf16)b.w;
    fr.e[8] = (__bf16)c.x;  fr.e[9] = (__bf16)c.y;  fr.e[10] = (__bf16)c.z; fr.e[11] = (__bf16)c.w;
    fr.e[12] = (__bf16)d.x; fr.e[13] = (__bf16)d.y; fr.e[14] = (__bf16)d.z; fr.e[15] = (__bf16)d.w;
    return fr.v;
}

__device__ __forceinline__ v16bf load_frag_bf16(const unsigned short* row, int k0) {
    Frag fr;
    fr.f4[0] = *(const float4*)(row + k0);
    fr.f4[1] = *(const float4*)(row + k0 + 16);
    return fr.v;
}

__device__ __forceinline__ v8f wmma_bf16(v16bf a, v16bf b, v8f c) {
    return __builtin_amdgcn_wmma_f32_16x16x32_bf16(false, a, false, b, (short)0, c,
                                                   false, false);
}

// Immediate-pattern lane swizzle (group-of-32 xor mode): 1 ds op, no index math.
#define SWZ_XOR(v, xm) \
    __int_as_float(__builtin_amdgcn_ds_swizzle(__float_as_int(v), ((xm) << 10) | 0x1f))

__device__ __forceinline__ float redmax16(float v) {
    v = fmaxf(v, SWZ_XOR(v, 1));
    v = fmaxf(v, SWZ_XOR(v, 2));
    v = fmaxf(v, SWZ_XOR(v, 4));
    v = fmaxf(v, SWZ_XOR(v, 8));
    return v;
}

// MODE 0: Q (RoPE + 1/sqrt(HD) fold, out [h][s][64]); MODE 1: K (RoPE); MODE 2: V ([h][64][s])
template <int MODE>
__global__ __launch_bounds__(256) void proj_rope_kernel(
    const float* __restrict__ X, const float* __restrict__ W,
    const float* __restrict__ bias, unsigned short* __restrict__ out)
{
    const int lane = threadIdx.x & 31;
    const int wave = threadIdx.x >> 5;
    const int m0 = blockIdx.x * 16;
    const int n0 = blockIdx.y * 128 + wave * 16;
    const int mrow = lane & 15, grp = lane >> 4;
    const int n = n0 + mrow;

    const float* xrow = X + (size_t)(m0 + mrow) * EMB;
    const float* wrow = W + (size_t)n * EMB;

    v8f acc = {};
    for (int kt = 0; kt < EMB; kt += 32) {
        v16bf a = load_frag_f32(xrow, kt + grp * 8);
        v16bf b = load_frag_f32(wrow, kt + grp * 8);
        acc = wmma_bf16(a, b, acc);
    }

    const float bn = bias[n];
    const int head = n >> 6, d = n & 63;
    float inv = 0.f;
    if constexpr (MODE < 2)
        inv = __powf(10000.f, -(float)(d & ~1) * (1.f / 64.f));

    #pragma unroll
    for (int r = 0; r < 8; ++r) {
        float x = acc[r] + bn;
        const int p = m0 + r + grp * 8;
        float y = x;
        if constexpr (MODE < 2) {
            float part = SWZ_XOR(x, 1);            // RoPE partner: feature d^1
            float ang = (float)p * inv, s, c;
            __sincosf(ang, &s, &c);
            y = (d & 1) ? (part * s + x * c) : (x * c - part * s);
        }
        if constexpr (MODE == 0) y *= 0.125f;      // fold softmax 1/sqrt(64)
        if constexpr (MODE < 2)
            out[(size_t)head * SEQ * HDIM + (size_t)p * HDIM + d] = f2bf(y);
        else
            out[(size_t)head * HDIM * SEQ + (size_t)d * SEQ + p] = f2bf(y);
    }
}

__global__ __launch_bounds__(256) void attn_kernel(
    const unsigned short* __restrict__ Qr, const unsigned short* __restrict__ Kr,
    const unsigned short* __restrict__ Vt, unsigned short* __restrict__ O)
{
    __shared__ unsigned short ldsP[8 * 16 * PSTRIDE];          // P spill tiles
    __shared__ unsigned short ldsV[8 * HDIM * PSTRIDE];        // async V tiles
    const int lane = threadIdx.x & 31;
    const int wave = threadIdx.x >> 5;
    const int head = blockIdx.y;
    const int q0 = blockIdx.x * 128 + wave * 16;
    const bool causal = head < CAUSAL_HEADS;
    const int mrow = lane & 15, grp = lane >> 4;
    unsigned short* P  = ldsP + wave * 16 * PSTRIDE;
    unsigned short* VB = ldsV + wave * HDIM * PSTRIDE;
    const unsigned vbase = (unsigned)(uintptr_t)VB;

    const unsigned short* Qh = Qr + (size_t)head * SEQ * HDIM;
    const unsigned short* Kh = Kr + (size_t)head * SEQ * HDIM;
    const unsigned short* Vh = Vt + (size_t)head * HDIM * SEQ;

    const unsigned short* qrow = Qh + (size_t)(q0 + mrow) * HDIM;
    v16bf qf0 = load_frag_bf16(qrow, grp * 8);
    v16bf qf1 = load_frag_bf16(qrow, 32 + grp * 8);

    Frag onef;                                   // all-ones B frag: row sums via WMMA
    #pragma unroll
    for (int i = 0; i < 16; ++i) onef.e[i] = (__bf16)1.0f;

    v8f acc[4] = {};
    float mrun[8], lrun[8];
    #pragma unroll
    for (int r = 0; r < 8; ++r) { mrun[r] = FLOORF; lrun[r] = 0.f; }

    const int jlo = causal ? 0
                           : (q0 - (WINDOW - 1) > 0 ? (((q0 - (WINDOW - 1)) >> 5) << 5) : 0);
    const int jhi = q0 + 16;   // exclusive; masking trims the rest

    for (int jt = jlo; jt < jhi; jt += 32) {
        // WAR guard on the single-buffered V tile (and P tile) from the
        // previous iteration, then kick off the async V copy so it overlaps
        // the QK WMMAs and softmax below.
        __asm volatile("s_wait_dscnt 0" ::: "memory");
        #pragma unroll
        for (int t = 0; t < 8; ++t) {
            const int c = t * 32 + lane;        // 256 chunks of 16B = 64x32 bf16
            const int d = c >> 2, co = c & 3;
            unsigned ldsaddr = vbase + (unsigned)(d * (PSTRIDE * 2) + co * 16);
            const void* gaddr = (const char*)Vh +
                                (((size_t)d * SEQ + jt) * 2 + (size_t)co * 16);
            __asm volatile("global_load_async_to_lds_b128 %0, %1, off"
                           :: "v"(ldsaddr), "v"(gaddr) : "memory");
        }
        if (jt + 32 < jhi)
            __builtin_prefetch(Kh + (size_t)(jt + 32 + mrow) * HDIM, 0, 1);

        // scores: 16 q-rows x 32 k-cols = two 16x16 D-frags, K-depth 64 = 2 WMMAs each
        v8f s0 = {}, s1 = {};
        {
            const unsigned short* k0row = Kh + (size_t)(jt + mrow) * HDIM;
            const unsigned short* k1row = Kh + (size_t)(jt + 16 + mrow) * HDIM;
            s0 = wmma_bf16(qf0, load_frag_bf16(k0row, grp * 8), s0);
            s0 = wmma_bf16(qf1, load_frag_bf16(k0row, 32 + grp * 8), s0);
            s1 = wmma_bf16(qf0, load_frag_bf16(k1row, grp * 8), s1);
            s1 = wmma_bf16(qf1, load_frag_bf16(k1row, 32 + grp * 8), s1);
        }

        const int j0 = jt + mrow, j1 = j0 + 16;
        float p0[8], p1[8], alpha[8];
        #pragma unroll
        for (int r = 0; r < 8; ++r) {
            const int i = q0 + r + grp * 8;
            // single-consumer predicates -> plain v_cmp + v_cndmask, no bool
            // materialization; masked lanes go to -inf and exp() kills them
            // (running max is clamped finite at FLOORF, so no NaN).
            float v0 = (causal ? (j0 <= i) : (j0 <= i && j0 >= i - (WINDOW - 1)))
                           ? s0[r] : -__builtin_inff();
            float v1 = (causal ? (j1 <= i) : (j1 <= i && j1 >= i - (WINDOW - 1)))
                           ? s1[r] : -__builtin_inff();
            float t = redmax16(fmaxf(v0, v1));
            float mn = fmaxf(mrun[r], t);
            alpha[r] = __expf(mrun[r] - mn);       // ==1 when unchanged
            mrun[r] = mn;
            p0[r] = __expf(v0 - mn);               // ==0 when masked
            p1[r] = __expf(v1 - mn);
        }
        #pragma unroll
        for (int dt = 0; dt < 4; ++dt)
            #pragma unroll
            for (int r = 0; r < 8; ++r) acc[dt][r] = acc[dt][r] * alpha[r];

        // spill P (16x32 bf16) to per-wave LDS tile for D->A fragment re-layout
        #pragma unroll
        for (int r = 0; r < 8; ++r) {
            const int m = r + grp * 8;
            P[m * PSTRIDE + mrow]      = f2bf(p0[r]);
            P[m * PSTRIDE + 16 + mrow] = f2bf(p1[r]);
        }
        __asm volatile("s_wait_dscnt 0" ::: "memory");

        v16bf pf;
        {
            Frag fr;
            const unsigned short* prow = P + mrow * PSTRIDE;
            fr.f4[0] = *(const float4*)(prow + grp * 8);
            fr.f4[1] = *(const float4*)(prow + 16 + grp * 8);
            pf = fr.v;
        }

        // row sums on the matrix core: D[m][*] = sum_k P[m][k]
        {
            v8f z = {};
            v8f rsum = wmma_bf16(pf, onef.v, z);
            #pragma unroll
            for (int r = 0; r < 8; ++r) lrun[r] = lrun[r] * alpha[r] + rsum[r];
        }

        // drain the async V tile, then P(16x32) x V(32x64) from LDS
        __asm volatile("s_wait_asynccnt 0" ::: "memory");
        #pragma unroll
        for (int dt = 0; dt < 4; ++dt) {
            const unsigned short* vrow = VB + (size_t)(dt * 16 + mrow) * PSTRIDE;
            Frag fr;
            fr.f4[0] = *(const float4*)(vrow + grp * 8);
            fr.f4[1] = *(const float4*)(vrow + 16 + grp * 8);
            acc[dt] = wmma_bf16(pf, fr.v, acc[dt]);
        }
    }

    #pragma unroll
    for (int dt = 0; dt < 4; ++dt)
        #pragma unroll
        for (int r = 0; r < 8; ++r) {
            const int p = q0 + r + grp * 8;
            float o = acc[dt][r] / lrun[r];
            O[(size_t)p * EMB + head * HDIM + dt * 16 + mrow] = f2bf(o);
        }
}

__global__ __launch_bounds__(256) void oproj_kernel(
    const unsigned short* __restrict__ A, const float* __restrict__ Wo,
    const float* __restrict__ bo, float* __restrict__ out)
{
    const int lane = threadIdx.x & 31;
    const int wave = threadIdx.x >> 5;
    const int m0 = blockIdx.x * 16;
    const int n0 = blockIdx.y * 128 + wave * 16;
    const int mrow = lane & 15, grp = lane >> 4;
    const int n = n0 + mrow;

    const unsigned short* arow = A + (size_t)(m0 + mrow) * EMB;
    const float* wrow = Wo + (size_t)n * EMB;

    v8f acc = {};
    for (int kt = 0; kt < EMB; kt += 32) {
        v16bf a = load_frag_bf16(arow, kt + grp * 8);
        v16bf b = load_frag_f32(wrow, kt + grp * 8);
        acc = wmma_bf16(a, b, acc);
    }
    const float bn = bo[n];
    #pragma unroll
    for (int r = 0; r < 8; ++r) {
        const int p = m0 + r + grp * 8;
        out[(size_t)p * EMB + n] = acc[r] + bn;
    }
}

extern "C" void kernel_launch(void* const* d_in, const int* in_sizes, int n_in,
                              void* d_out, int out_size, void* d_ws, size_t ws_size,
                              hipStream_t stream) {
    const float* query = (const float*)d_in[0];
    const float* key   = (const float*)d_in[1];
    const float* value = (const float*)d_in[2];
    const float* Wq = (const float*)d_in[3];
    const float* bq = (const float*)d_in[4];
    const float* Wk = (const float*)d_in[5];
    const float* bk = (const float*)d_in[6];
    const float* Wv = (const float*)d_in[7];
    const float* bv = (const float*)d_in[8];
    const float* Wo = (const float*)d_in[9];
    const float* bo = (const float*)d_in[10];

    unsigned short* Qr = (unsigned short*)d_ws;            // [H][S][64] bf16 (pre-scaled)
    unsigned short* Kr = Qr + (size_t)SEQ * EMB;           // [H][S][64] bf16
    unsigned short* Vt = Kr + (size_t)SEQ * EMB;           // [H][64][S] bf16
    unsigned short* At = Vt + (size_t)SEQ * EMB;           // [S][E]     bf16

    dim3 gproj(SEQ / 16, EMB / 128);
    proj_rope_kernel<0><<<gproj, 256, 0, stream>>>(query, Wq, bq, Qr);
    proj_rope_kernel<1><<<gproj, 256, 0, stream>>>(key,   Wk, bk, Kr);
    proj_rope_kernel<2><<<gproj, 256, 0, stream>>>(value, Wv, bv, Vt);

    attn_kernel<<<dim3(SEQ / 128, NH), 256, 0, stream>>>(Qr, Kr, Vt, At);

    oproj_kernel<<<gproj, 256, 0, stream>>>(At, Wo, bo, (float*)d_out);
}
